// SwinBlock_74887049773761
// MI455X (gfx1250) — compile-verified, matrix operand order checked by reading, not run on hardware
//
#include <hip/hip_runtime.h>
#include <hip/hip_bf16.h>
#include <math.h>

typedef __attribute__((ext_vector_type(16))) __bf16 v16bf;
typedef __attribute__((ext_vector_type(8)))  float  v8f;
typedef __attribute__((ext_vector_type(4)))  int    v4i;
typedef unsigned short u16;
typedef unsigned int   u32;

#define DIM   384
#define NHEAD 12
#define HD    32
#define NTOK  32768          // 8 images * 64*64 tokens
#define SCALE 0.05103103630798287f   // 384^-0.5 (reference scales by full C)

#define HAS_ASYNC_LDS __has_builtin(__builtin_amdgcn_global_load_async_to_lds_b128)

__device__ __forceinline__ u16 f2bf(float f) {
  u32 u = __float_as_uint(f);
  u += 0x7FFFu + ((u >> 16) & 1u);        // round-to-nearest-even
  return (u16)(u >> 16);
}

// window-order token index -> original token index (window partition/reverse)
__device__ __forceinline__ int win2orig(int t) {
  int b = t >> 12;
  int r = t & 4095;
  int w = r >> 6;           // window within image (wy*8+wx)
  int p = r & 63;           // token within window (py*8+px)
  int row = ((w >> 3) << 3) + (p >> 3);
  int col = ((w & 7) << 3) + (p & 7);
  return (b << 12) + row * 64 + col;
}

union Frag16 { v16bf v; u16 u[16]; };

#if HAS_ASYNC_LDS
// per-lane async 16-byte copy global -> LDS, tracked by ASYNCcnt
__device__ __forceinline__ void async_cp16(const void* g, void* l) {
  __builtin_amdgcn_global_load_async_to_lds_b128(
      (__attribute__((address_space(1))) v4i*)g,
      (__attribute__((address_space(3))) v4i*)l, 0, 0);
}
template <int N> __device__ __forceinline__ void async_wait() {
#if __has_builtin(__builtin_amdgcn_s_wait_asynccnt)
  __builtin_amdgcn_s_wait_asynccnt(N);
#else
  asm volatile("s_wait_asynccnt %0" :: "i"(N) : "memory");
#endif
}
#endif

// ---------------------------------------------------------------- weights
__global__ void cvt_bf16_kernel(const float* __restrict__ in, u16* __restrict__ out, int n) {
  int i = blockIdx.x * 256 + threadIdx.x;
  if (i < n) out[i] = f2bf(in[i]);
}

// ---------------------------------------------------------------- LayerNorm
// one token per block, 128 threads, 3 channels/thread; optional windowed gather
__global__ void ln_kernel(const float* __restrict__ in, const float* __restrict__ w,
                          const float* __restrict__ b, u16* __restrict__ out, int gather) {
  __shared__ float red[128];
  int t = blockIdx.x, tid = threadIdx.x;
  int src = gather ? win2orig(t) : t;
  const float* row = in + (size_t)src * DIM;
  float xv[3]; float s = 0.f;
#pragma unroll
  for (int i = 0; i < 3; ++i) { xv[i] = row[tid + i * 128]; s += xv[i]; }
  red[tid] = s; __syncthreads();
  for (int off = 64; off > 0; off >>= 1) { if (tid < off) red[tid] += red[tid + off]; __syncthreads(); }
  float mu = red[0] * (1.f / DIM);
  __syncthreads();
  float s2 = 0.f;
#pragma unroll
  for (int i = 0; i < 3; ++i) { float d = xv[i] - mu; s2 += d * d; }
  red[tid] = s2; __syncthreads();
  for (int off = 64; off > 0; off >>= 1) { if (tid < off) red[tid] += red[tid + off]; __syncthreads(); }
  float inv = rsqrtf(red[0] * (1.f / DIM) + 1e-5f);
  u16* orow = out + (size_t)t * DIM;
#pragma unroll
  for (int i = 0; i < 3; ++i) {
    int c = tid + i * 128;
    orow[c] = f2bf((xv[i] - mu) * inv * w[c] + b[c]);
  }
}

// ---------------------------------------------------------------- WMMA GEMM
// C[M,N] = A[M,K](bf16) @ B[K,N](bf16) + bias, block tile 128x128, 8 waves,
// wave tile 64x32 (4x2 of 16x16 C frags), K-step 32 via v_wmma_f32_16x16x32_bf16.
// Double-buffered LDS tiles staged with GLOBAL_LOAD_ASYNC_TO_LDS_B128 (ASYNCcnt)
// when the toolchain exposes the builtin; synchronous staging otherwise.
// EPI: 0 = bf16 out; 1 = exact GELU -> bf16; 2 = +gathered-x residual -> f32;
//      3 = +residual -> f32 scattered with window-reverse (final output).
template <int EPI>
__launch_bounds__(256)
__global__ void gemm_bf16_kernel(const u16* __restrict__ A, const u16* __restrict__ B,
                                 const float* __restrict__ bias,
                                 u16* __restrict__ outB, float* __restrict__ outF,
                                 const float* __restrict__ resid,
                                 int M, int N, int K) {
#if HAS_ASYNC_LDS
  __shared__ u16 As[2][128 * 32];
  __shared__ u16 Bs[2][32 * 128];
#else
  __shared__ u16 As[1][128 * 32];
  __shared__ u16 Bs[1][32 * 128];
#endif
  int tid = threadIdx.x;
  int lane = tid & 31, wv = tid >> 5;
  int half = lane >> 4, l16 = lane & 15;
  int wm = wv >> 2, wn = wv & 3;
  int m0 = blockIdx.y * 128, n0 = blockIdx.x * 128;

  v8f acc[4][2] = {};
  int nk = K >> 5;

  auto compute_tile = [&](const u16* as, const u16* bs) {
    Frag16 a[4], b[2];
#pragma unroll
    for (int tm = 0; tm < 4; ++tm) {
      int m = wm * 64 + tm * 16 + l16;
#pragma unroll
      for (int p = 0; p < 8; ++p) {
        int k = 2 * p + half * 8 + (p >= 4 ? 8 : 0);   // CDNA5 16-bit A layout
        a[tm].u[2 * p]     = as[m * 32 + k];
        a[tm].u[2 * p + 1] = as[m * 32 + k + 1];
      }
    }
#pragma unroll
    for (int tn = 0; tn < 2; ++tn) {
      int n = wn * 32 + tn * 16 + l16;
#pragma unroll
      for (int p = 0; p < 8; ++p) {
        int k = 2 * p + half * 8 + (p >= 4 ? 8 : 0);
        b[tn].u[2 * p]     = bs[k * 128 + n];
        b[tn].u[2 * p + 1] = bs[(k + 1) * 128 + n];
      }
    }
#pragma unroll
    for (int tm = 0; tm < 4; ++tm)
#pragma unroll
      for (int tn = 0; tn < 2; ++tn)
        acc[tm][tn] = __builtin_amdgcn_wmma_f32_16x16x32_bf16(
            false, a[tm].v, false, b[tn].v, (short)0, acc[tm][tn], false, false);
  };

#if HAS_ASYNC_LDS
  auto stage = [&](int buf, int k0) {
#pragma unroll
    for (int it = 0; it < 2; ++it) {
      int ch = it * 256 + tid;                 // 512 x 16B chunks per tile
      { int r = ch >> 2, c8 = (ch & 3) << 3;   // A: 128 rows x 32
        async_cp16(A + (size_t)(m0 + r) * K + k0 + c8, &As[buf][ch * 8]); }
      { int r = ch >> 4, c8 = (ch & 15) << 3;  // B: 32 rows x 128
        async_cp16(B + (size_t)(k0 + r) * N + n0 + c8, &Bs[buf][ch * 8]); }
    }
  };
  stage(0, 0);
  for (int kt = 0; kt < nk; ++kt) {
    int cur = kt & 1;
    if (kt + 1 < nk) {
      stage(cur ^ 1, (kt + 1) << 5);  // next tile in flight...
      async_wait<4>();                // ...while current stage (4 older ops) lands
    } else {
      async_wait<0>();
    }
    __syncthreads();                  // every wave waited its own copies -> tile ready
    compute_tile(As[cur], Bs[cur]);
    __syncthreads();                  // protect buffer from stage kt+2 overwrite
  }
#else
  for (int kt = 0; kt < nk; ++kt) {
    int k0 = kt << 5;
    __syncthreads();
#pragma unroll
    for (int it = 0; it < 2; ++it) {
      int ch = it * 256 + tid;
      { int r = ch >> 2, c8 = (ch & 3) << 3;
        ((uint4*)As[0])[ch] = *(const uint4*)(A + (size_t)(m0 + r) * K + k0 + c8); }
      { int r = ch >> 4, c8 = (ch & 15) << 3;
        ((uint4*)Bs[0])[ch] = *(const uint4*)(B + (size_t)(k0 + r) * N + n0 + c8); }
    }
    if (kt + 1 < nk)
      __builtin_prefetch(A + (size_t)(m0 + (tid >> 1)) * K + k0 + 32, 0, 1);
    __syncthreads();
    compute_tile(As[0], Bs[0]);
  }
#endif

#pragma unroll
  for (int tm = 0; tm < 4; ++tm)
#pragma unroll
    for (int tn = 0; tn < 2; ++tn)
#pragma unroll
      for (int r = 0; r < 8; ++r) {            // C layout: VGPR r -> M=r / r+8
        int mg = m0 + wm * 64 + tm * 16 + half * 8 + r;
        int ng = n0 + wn * 32 + tn * 16 + l16;
        float v = acc[tm][tn][r] + bias[ng];
        if constexpr (EPI == 0) {
          outB[(size_t)mg * N + ng] = f2bf(v);
        } else if constexpr (EPI == 1) {
          v = 0.5f * v * (1.f + erff(v * 0.70710678118654752f));
          outB[(size_t)mg * N + ng] = f2bf(v);
        } else if constexpr (EPI == 2) {
          v += resid[(size_t)win2orig(mg) * DIM + ng];   // residual = window-gathered x
          outF[(size_t)mg * DIM + ng] = v;
        } else {
          v += resid[(size_t)mg * DIM + ng];             // residual = xw2
          outF[(size_t)win2orig(mg) * DIM + ng] = v;     // window reverse
        }
      }
}

// ---------------------------------------------------------------- attention
// one (window, head) per block, 128 threads = 4 waves.
// S = q@k^T (WMMA K=32) -> LDS softmax (+rel bias, closed-form index) -> P@V (WMMA K=64)
__global__ void attn_kernel(const u16* __restrict__ qkv, const float* __restrict__ btab,
                            u16* __restrict__ out) {
  __shared__ u16 qs[64 * 32], ks_[64 * 32], vs[64 * 32];
  __shared__ float S[64 * 64];
  __shared__ u16 Ps[64 * 64];
  int head = blockIdx.x, win = blockIdx.y, base = win * 64;
  int tid = threadIdx.x;
  int lane = tid & 31, wv = tid >> 5;
  int half = lane >> 4, l16 = lane & 15;

#pragma unroll
  for (int it = 0; it < 2; ++it) {
    int ch = it * 128 + tid;                  // 256 uint4 chunks per matrix
    int r = ch >> 2, c8 = (ch & 3) << 3;
    size_t o = (size_t)(base + r) * 1152 + head * 32 + c8;
    ((uint4*)qs)[ch]  = *(const uint4*)(qkv + o);
    ((uint4*)ks_)[ch] = *(const uint4*)(qkv + o + 384);
    ((uint4*)vs)[ch]  = *(const uint4*)(qkv + o + 768);
  }
  __syncthreads();

  int r0 = wv * 16;
  Frag16 a;
#pragma unroll
  for (int p = 0; p < 8; ++p) {
    int k = 2 * p + half * 8 + (p >= 4 ? 8 : 0);
    int m = r0 + l16;
    a.u[2 * p]     = qs[m * 32 + k];
    a.u[2 * p + 1] = qs[m * 32 + k + 1];
  }
  v8f sacc[4] = {};
#pragma unroll
  for (int tn = 0; tn < 4; ++tn) {
    Frag16 b;
    int n = tn * 16 + l16;
#pragma unroll
    for (int p = 0; p < 8; ++p) {
      int k = 2 * p + half * 8 + (p >= 4 ? 8 : 0);
      b.u[2 * p]     = ks_[n * 32 + k];        // k^T: B(k=d, n=col) = K[col][d]
      b.u[2 * p + 1] = ks_[n * 32 + k + 1];
    }
    sacc[tn] = __builtin_amdgcn_wmma_f32_16x16x32_bf16(
        false, a.v, false, b.v, (short)0, sacc[tn], false, false);
  }
#pragma unroll
  for (int tn = 0; tn < 4; ++tn)
#pragma unroll
    for (int r = 0; r < 8; ++r)
      S[(r0 + half * 8 + r) * 64 + tn * 16 + l16] = sacc[tn][r];
  __syncthreads();

  if (tid < 64) {                              // softmax, one row per thread
    int row = tid;
    float mx = -3.4e38f;
    for (int c = 0; c < 64; ++c) {
      int dy = (row >> 3) - (c >> 3) + 7;
      int dx = (row & 7) - (c & 7) + 7;
      float v = S[row * 64 + c] * SCALE + btab[(dy * 15 + dx) * NHEAD + head];
      S[row * 64 + c] = v;
      mx = fmaxf(mx, v);
    }
    float sum = 0.f;
    for (int c = 0; c < 64; ++c) {
      float e = __expf(S[row * 64 + c] - mx);
      S[row * 64 + c] = e;
      sum += e;
    }
    float inv = 1.f / sum;
    for (int c = 0; c < 64; ++c) Ps[row * 64 + c] = f2bf(S[row * 64 + c] * inv);
  }
  __syncthreads();

  v8f oacc[2] = {};
#pragma unroll
  for (int kk = 0; kk < 64; kk += 32) {
    Frag16 pa;
#pragma unroll
    for (int p = 0; p < 8; ++p) {
      int k = kk + 2 * p + half * 8 + (p >= 4 ? 8 : 0);
      int m = r0 + l16;
      pa.u[2 * p]     = Ps[m * 64 + k];
      pa.u[2 * p + 1] = Ps[m * 64 + k + 1];
    }
#pragma unroll
    for (int tn = 0; tn < 2; ++tn) {
      Frag16 b;
      int n = tn * 16 + l16;
#pragma unroll
      for (int p = 0; p < 8; ++p) {
        int k = kk + 2 * p + half * 8 + (p >= 4 ? 8 : 0);
        b.u[2 * p]     = vs[k * 32 + n];
        b.u[2 * p + 1] = vs[(k + 1) * 32 + n];
      }
      oacc[tn] = __builtin_amdgcn_wmma_f32_16x16x32_bf16(
          false, pa.v, false, b.v, (short)0, oacc[tn], false, false);
    }
  }
#pragma unroll
  for (int tn = 0; tn < 2; ++tn)
#pragma unroll
    for (int r = 0; r < 8; ++r) {
      int row = r0 + half * 8 + r;
      int col = tn * 16 + l16;
      out[(size_t)(base + row) * DIM + head * 32 + col] = f2bf(oacc[tn][r]);
    }
}

// ---------------------------------------------------------------- launch
extern "C" void kernel_launch(void* const* d_in, const int* in_sizes, int n_in,
                              void* d_out, int out_size, void* d_ws, size_t ws_size,
                              hipStream_t stream) {
  const float* x      = (const float*)d_in[0];
  const float* ln1_w  = (const float*)d_in[1];
  const float* ln1_b  = (const float*)d_in[2];
  const float* qkv_w  = (const float*)d_in[3];
  const float* qkv_b  = (const float*)d_in[4];
  const float* btab   = (const float*)d_in[5];
  const float* proj_w = (const float*)d_in[6];
  const float* proj_b = (const float*)d_in[7];
  const float* ln2_w  = (const float*)d_in[8];
  const float* ln2_b  = (const float*)d_in[9];
  const float* fc1_w  = (const float*)d_in[10];
  const float* fc1_b  = (const float*)d_in[11];
  const float* fc2_w  = (const float*)d_in[12];
  const float* fc2_b  = (const float*)d_in[13];
  float* out = (float*)d_out;
  (void)in_sizes; (void)n_in; (void)out_size; (void)ws_size;

  char* ws = (char*)d_ws;
  size_t off = 0;
  auto alloc = [&](size_t bytes) -> void* {
    void* p = ws + off;
    off += (bytes + 255) & ~(size_t)255;
    return p;
  };
  u16*   wqkv  = (u16*)alloc((size_t)384 * 1152 * 2);
  u16*   wproj = (u16*)alloc((size_t)384 * 384 * 2);
  u16*   wfc1  = (u16*)alloc((size_t)384 * 1536 * 2);
  u16*   wfc2  = (u16*)alloc((size_t)1536 * 384 * 2);
  u16*   act   = (u16*)alloc((size_t)NTOK * DIM * 2);     // h -> attn_out -> h2
  u16*   big   = (u16*)alloc((size_t)NTOK * 1536 * 2);    // qkv -> mlp hidden
  float* xw2   = (float*)alloc((size_t)NTOK * DIM * 4);   // post-attention residual

  cvt_bf16_kernel<<<(384 * 1152 + 255) / 256, 256, 0, stream>>>(qkv_w, wqkv, 384 * 1152);
  cvt_bf16_kernel<<<(384 * 384  + 255) / 256, 256, 0, stream>>>(proj_w, wproj, 384 * 384);
  cvt_bf16_kernel<<<(384 * 1536 + 255) / 256, 256, 0, stream>>>(fc1_w, wfc1, 384 * 1536);
  cvt_bf16_kernel<<<(1536 * 384 + 255) / 256, 256, 0, stream>>>(fc2_w, wfc2, 1536 * 384);

  ln_kernel<<<NTOK, 128, 0, stream>>>(x, ln1_w, ln1_b, act, 1);
  gemm_bf16_kernel<0><<<dim3(1152 / 128, NTOK / 128), 256, 0, stream>>>(
      act, wqkv, qkv_b, big, nullptr, nullptr, NTOK, 1152, 384);
  attn_kernel<<<dim3(NHEAD, 512), 128, 0, stream>>>(big, btab, act);
  gemm_bf16_kernel<2><<<dim3(384 / 128, NTOK / 128), 256, 0, stream>>>(
      act, wproj, proj_b, nullptr, xw2, x, NTOK, 384, 384);
  ln_kernel<<<NTOK, 128, 0, stream>>>(xw2, ln2_w, ln2_b, act, 0);
  gemm_bf16_kernel<1><<<dim3(1536 / 128, NTOK / 128), 256, 0, stream>>>(
      act, wfc1, fc1_b, big, nullptr, nullptr, NTOK, 1536, 384);
  gemm_bf16_kernel<3><<<dim3(384 / 128, NTOK / 128), 256, 0, stream>>>(
      big, wfc2, fc2_b, nullptr, out, xw2, NTOK, 384, 1536);
}
